// RelationalSelfAttention_54623394070799
// MI455X (gfx1250) — matrix operand order repaired and checked by reading, not compiled, not run on hardware
//
#include <hip/hip_runtime.h>
#include <math.h>

typedef __attribute__((ext_vector_type(16))) _Float16 v16h;
typedef __attribute__((ext_vector_type(8)))  float    v8f;

constexpr int Bn = 32, Nn = 512, Dn = 256, Hn = 8, DHn = 32;
constexpr float EPSV = 1e-5f;
constexpr float LNEG = -1.7014118e38f;   // finfo(f32).min / 2

// ---------------- WMMA fragment helpers (CDNA5 16x16x32 f16 layouts) -------

// A-matrix 16x32 f16: lane m = lane&15, group g = lane>>4.
// element i:  k = 16*g + i   (contiguous, matches B fragments)
__device__ __forceinline__ v16h load_a_frag(const _Float16* rowptr, int g) {
  v16h a;
#pragma unroll
  for (int v = 0; v < 8; ++v) {
    int k0 = g * 16 + v * 2;
    a[2 * v]     = rowptr[k0];
    a[2 * v + 1] = rowptr[k0 + 1];
  }
  return a;
}

// B-matrix 32x16 f16: lane n = lane&15, group g = lane>>4 holds K = g*16..g*16+15,
// i.e. 16 contiguous halves per lane when storage is K-contiguous per column.
__device__ __forceinline__ v16h load_b_frag(const _Float16* p) {
  v16h b;
#pragma unroll
  for (int i = 0; i < 16; ++i) b[i] = p[i];
  return b;
}

__device__ __forceinline__ float wave_max(float v) {
#pragma unroll
  for (int off = 16; off > 0; off >>= 1) v = fmaxf(v, __shfl_xor(v, off, 32));
  return v;
}
__device__ __forceinline__ float wave_sum(float v) {
#pragma unroll
  for (int off = 16; off > 0; off >>= 1) v += __shfl_xor(v, off, 32);
  return v;
}

// ---- CDNA5 async DMA global->LDS helpers (ASYNCcnt-tracked, bypass VGPRs) --
__device__ __forceinline__ void async_g2l_b128(unsigned lds_off, const void* gptr) {
  unsigned long long ga = (unsigned long long)(size_t)gptr;
  asm volatile("global_load_async_to_lds_b128 %0, %1, off"
               :: "v"(lds_off), "v"(ga) : "memory");
}
__device__ __forceinline__ void async_g2l_b64(unsigned lds_off, const void* gptr) {
  unsigned long long ga = (unsigned long long)(size_t)gptr;
  asm volatile("global_load_async_to_lds_b64 %0, %1, off"
               :: "v"(lds_off), "v"(ga) : "memory");
}
__device__ __forceinline__ void wait_async0() {
  asm volatile("s_wait_asynccnt 0" ::: "memory");
}
__device__ __forceinline__ unsigned lds_addr_of(const void* p) {
  // LDS aperture: addr[31:0] is the LDS byte address (ISA 10.2)
  return (unsigned)(size_t)p;
}

// ---------------- Kernel 0: x -> f16 copy ----------------------------------
__global__ void k_xh(const float* __restrict__ x, _Float16* __restrict__ xh) {
  int idx = blockIdx.x * blockDim.x + threadIdx.x;
  const int total = Bn * Nn * Dn;
  for (; idx < total; idx += gridDim.x * blockDim.x) xh[idx] = (_Float16)x[idx];
}

// ---------------- Kernel 1: weight swizzle (f16) + heading precompute ------
// WqkvS: [kc(8)][nt(48)][lane(32)][h(16)] halves; element = Wqkv[k][n],
//   k = kc*32 + (lane>>4)*16 + h,  n = nt*16 + (lane&15)   (B-fragment major)
__global__ void k_prep(const float* __restrict__ Wqkv, const float* __restrict__ Wout,
                       const float* __restrict__ att,
                       _Float16* __restrict__ WqkvS, _Float16* __restrict__ WoutS,
                       float* __restrict__ hdg) {
  int idx = blockIdx.x * blockDim.x + threadIdx.x;
  const int total = 196608 + 65536 + Bn * Nn;
  for (; idx < total; idx += gridDim.x * blockDim.x) {
    if (idx < 196608) {
      int h = idx & 15, lane = (idx >> 4) & 31, tile = idx >> 9;
      int nt = tile % 48, kc = tile / 48;
      int k = kc * 32 + (lane >> 4) * 16 + h;
      int n = nt * 16 + (lane & 15);
      WqkvS[idx] = (_Float16)Wqkv[k * 768 + n];
    } else if (idx < 262144) {
      int j = idx - 196608;
      int h = j & 15, lane = (j >> 4) & 31, tile = j >> 9;
      int nt = tile & 15, kc = tile >> 4;
      int k = kc * 32 + (lane >> 4) * 16 + h;
      int n = nt * 16 + (lane & 15);
      WoutS[j] = (_Float16)Wout[k * 256 + n];
    } else {
      int j = idx - 262144;
      float a = att[j];
      hdg[2 * j]     = cosf(a);
      hdg[2 * j + 1] = sinf(a);
    }
  }
}

// ---------------- Kernel 2: QKV projection (WMMA, async-LDS staging) -------
// grid = B * (N/16); one 16-row tile of X @ Wqkv(256x768).
__global__ void k_qkv(const _Float16* __restrict__ xh, const _Float16* __restrict__ WqkvS,
                      _Float16* __restrict__ qh, _Float16* __restrict__ kh,
                      _Float16* __restrict__ vt) {
  __shared__ _Float16 Xs[16][256];   // 8 KB, contiguous image of 16 rows of xh
  int b  = blockIdx.x >> 5;
  int i0 = (blockIdx.x & 31) * 16;
  int tid = threadIdx.x;

  // async DMA the 8 KB tile into LDS (two b128 per thread)
  {
    const char* src = (const char*)(xh + ((size_t)(b * Nn) + i0) * Dn);
    unsigned base = lds_addr_of(&Xs[0][0]);
    async_g2l_b128(base + tid * 16,        src + tid * 16);
    async_g2l_b128(base + 4096 + tid * 16, src + 4096 + tid * 16);
  }
  wait_async0();
  __syncthreads();

  int wave = tid >> 5, lane = tid & 31, g = lane >> 4, l15 = lane & 15;

  // hoist all A fragments (K = 256 -> 8 chunks) into registers
  v16h afr[8];
#pragma unroll
  for (int kc = 0; kc < 8; ++kc) afr[kc] = load_a_frag(&Xs[l15][kc * 32], g);

#pragma unroll 1
  for (int t = 0; t < 6; ++t) {
    int nt = wave * 6 + t;
    const _Float16* bptr = WqkvS + (size_t)nt * 512 + lane * 16;
    if (t < 5) __builtin_prefetch(bptr + 512, 0, 0);  // next nt -> global_prefetch_b8
    v8f acc = {};
    v16h bcur = load_b_frag(bptr);
#pragma unroll
    for (int kc = 0; kc < 8; ++kc) {                  // double-buffered B stream
      v16h bnxt = bcur;
      if (kc < 7) bnxt = load_b_frag(bptr + (size_t)(kc + 1) * 48 * 512);
      acc = __builtin_amdgcn_wmma_f32_16x16x32_f16(false, afr[kc], false, bcur,
                                                   (short)0, acc, false, false);
      bcur = bnxt;
    }
#pragma unroll
    for (int r = 0; r < 8; ++r) {
      int m  = r + 8 * g;
      int gi = i0 + m;
      int c  = nt * 16 + l15;
      _Float16 val = (_Float16)acc[r];
      if (c < 256) {                       // Q -> (B,H,N,DH)
        int h = c >> 5, dh = c & 31;
        qh[(((size_t)(b * Hn + h) * Nn) + gi) * DHn + dh] = val;
      } else if (c < 512) {                // K -> (B,H,N,DH)
        int c2 = c - 256, h = c2 >> 5, dh = c2 & 31;
        kh[(((size_t)(b * Hn + h) * Nn) + gi) * DHn + dh] = val;
      } else {                             // V -> transposed (B,H,DH,N)
        int c2 = c - 512, h = c2 >> 5, dh = c2 & 31;
        vt[(((size_t)(b * Hn + h) * DHn) + dh) * Nn + gi] = val;
      }
    }
  }
}

// ---------------- Kernel 3: fused attention (scores+bias+softmax+PV) -------
// grid = B*H*(N/32); dynamic LDS: S[32][512] f32, P[32][512] f16, Qs, row geo.
__global__ void k_attn(const _Float16* __restrict__ qh, const _Float16* __restrict__ kh,
                       const _Float16* __restrict__ vt,
                       const float* __restrict__ pos, const float* __restrict__ vel,
                       const float* __restrict__ hdg,
                       const float* __restrict__ relW, const float* __restrict__ relb,
                       const int* __restrict__ alive, _Float16* __restrict__ ao) {
  extern __shared__ char smem[];
  float    (*S)[512]  = (float(*)[512])smem;                          // 64 KB
  _Float16 (*P)[512]  = (_Float16(*)[512])(smem + 65536);             // 32 KB
  _Float16 (*Qs)[32]  = (_Float16(*)[32])(smem + 65536 + 32768);      //  2 KB
  float*   rowGeo     = (float*)(smem + 65536 + 32768 + 2048);        // 32*6 f32
  float*   wcol       = rowGeo + 192;                                 // 7 f32

  int bh = blockIdx.x >> 4;
  int i0 = (blockIdx.x & 15) * 32;
  int b = bh >> 3, h = bh & 7;
  int tid = threadIdx.x, wave = tid >> 5, lane = tid & 31, g = lane >> 4, l15 = lane & 15;

  // async DMA the 2 KB q tile (rows i0..i0+31 are contiguous in qh)
  {
    const char* src = (const char*)(qh + ((size_t)(bh * Nn) + i0) * DHn);
    async_g2l_b64(lds_addr_of(&Qs[0][0]) + tid * 8, src + tid * 8);
  }
  if (tid < 32) {
    int gi = b * Nn + i0 + tid;
    rowGeo[tid * 6 + 0] = pos[gi * 2 + 0];
    rowGeo[tid * 6 + 1] = pos[gi * 2 + 1];
    rowGeo[tid * 6 + 2] = vel[gi * 2 + 0];
    rowGeo[tid * 6 + 3] = vel[gi * 2 + 1];
    rowGeo[tid * 6 + 4] = hdg[gi * 2 + 0];
    rowGeo[tid * 6 + 5] = hdg[gi * 2 + 1];
  }
  if (tid < 6)  wcol[tid] = relW[tid * Hn + h];
  if (tid == 6) wcol[6]   = relb[h];
  wait_async0();
  __syncthreads();

  // ---- scores: 2 m-tiles x 32 j-tiles, one WMMA each (K = DH = 32),
  //      double-buffered K-fragment stream ----
  {
    const _Float16* kbase = kh + ((size_t)(bh * Nn) + l15) * DHn + g * 16;
    v16h bcur = load_b_frag(kbase + (size_t)((wave * 8) & 31) * 16 * DHn);
#pragma unroll 1
    for (int s = 0; s < 8; ++s) {
      int tt = wave * 8 + s;
      int mt = tt >> 5, jt = tt & 31;
      v16h bnxt = bcur;
      if (s < 7) bnxt = load_b_frag(kbase + (size_t)((tt + 1) & 31) * 16 * DHn);
      v16h a = load_a_frag(&Qs[mt * 16 + l15][0], g);
      v8f acc = {};
      acc = __builtin_amdgcn_wmma_f32_16x16x32_f16(false, a, false, bcur,
                                                   (short)0, acc, false, false);
#pragma unroll
      for (int r = 0; r < 8; ++r) S[mt * 16 + r + 8 * g][jt * 16 + l15] = acc[r];
      bcur = bnxt;
    }
  }
  __syncthreads();

  // ---- scale + relational bias + key mask (fp32 VALU, never hits HBM) ----
  const float scale = 0.17677669529663687f;  // 1/sqrt(32)
  for (int idx = tid; idx < 32 * 512; idx += 256) {
    int i = idx >> 9, j = idx & 511;
    int gj = b * Nn + j;
    float dpx = pos[gj * 2]     - rowGeo[i * 6 + 0];
    float dpy = pos[gj * 2 + 1] - rowGeo[i * 6 + 1];
    float dist = sqrtf(dpx * dpx + dpy * dpy + 1e-8f);
    float inv  = 1.0f / dist;
    float dvx = vel[gj * 2]     - rowGeo[i * 6 + 2];
    float dvy = vel[gj * 2 + 1] - rowGeo[i * 6 + 3];
    float align = hdg[gj * 2] * rowGeo[i * 6 + 4] + hdg[gj * 2 + 1] * rowGeo[i * 6 + 5];
    float bias = log1pf(dist) * wcol[0] + dpx * inv * wcol[1] + dpy * inv * wcol[2]
               + dvx * wcol[3] + dvy * wcol[4] + align * wcol[5] + wcol[6];
    float maskt = alive[gj] ? 0.0f : LNEG;
    S[i][j] = S[i][j] * scale + bias + maskt;
  }
  __syncthreads();

  // ---- row softmax: 4 rows per wave, shuffle reductions (wave32) ----
  for (int rr = 0; rr < 4; ++rr) {
    int i = wave * 4 + rr;
    float m = -3.4e38f;
    for (int c = lane; c < 512; c += 32) m = fmaxf(m, S[i][c]);
    m = wave_max(m);
    float sum = 0.f;
    for (int c = lane; c < 512; c += 32) { float e = __expf(S[i][c] - m); S[i][c] = e; sum += e; }
    sum = wave_sum(sum);
    float invs = 1.0f / sum;
    for (int c = lane; c < 512; c += 32) P[i][c] = (_Float16)(S[i][c] * invs);
  }
  __syncthreads();

  // ---- PV: out(32x32) = P(32x512) @ V(512x32); all 8 waves busy:
  //      4 C-tiles x 2 K-halves, partials reduced through LDS ----
  {
    int t = wave & 3, half = wave >> 2;
    int mt = t >> 1, nt = t & 1;
    const _Float16* vbase = vt + ((size_t)(bh * DHn) + nt * 16 + l15) * Nn + g * 16;
    int jc0 = half * 8;
    v8f acc = {};
    v16h bcur = load_b_frag(vbase + (size_t)jc0 * 32);
#pragma unroll
    for (int jj = 0; jj < 8; ++jj) {
      int jc = jc0 + jj;
      v16h bnxt = bcur;
      if (jj < 7) bnxt = load_b_frag(vbase + (size_t)(jc + 1) * 32);
      v16h a = load_a_frag(&P[mt * 16 + l15][jc * 32], g);
      acc = __builtin_amdgcn_wmma_f32_16x16x32_f16(false, a, false, bcur,
                                                   (short)0, acc, false, false);
      bcur = bnxt;
    }
    float* Sf = (float*)smem;   // scores LDS is free now: 8 KB partial scratch
#pragma unroll
    for (int r = 0; r < 8; ++r) Sf[(t * 2 + half) * 256 + r * 32 + lane] = acc[r];
  }
  __syncthreads();
  if (wave < 4) {
    int t = wave, mt = t >> 1, nt = t & 1;
    const float* Sf = (const float*)smem;
#pragma unroll
    for (int r = 0; r < 8; ++r) {
      float v0 = Sf[(t * 2 + 0) * 256 + r * 32 + lane]
               + Sf[(t * 2 + 1) * 256 + r * 32 + lane];
      int row = i0 + mt * 16 + r + 8 * g;
      int col = h * DHn + nt * 16 + l15;
      ao[((size_t)(b * Nn) + row) * Dn + col] = (_Float16)v0;
    }
  }
}

// ---------------- Kernel 4: output projection + residual + LayerNorm -------
// grid = B*(N/16); Y(16x256) = AO(16x256)@Wout, then per-row LN fused.
__global__ void k_proj(const _Float16* __restrict__ ao, const _Float16* __restrict__ WoutS,
                       const float* __restrict__ x, const float* __restrict__ gamma,
                       const float* __restrict__ beta, float* __restrict__ out) {
  __shared__ _Float16 AOs[16][256];
  __shared__ float Y[16][256];
  int b  = blockIdx.x >> 5;
  int i0 = (blockIdx.x & 31) * 16;
  int tid = threadIdx.x, wave = tid >> 5, lane = tid & 31, g = lane >> 4, l15 = lane & 15;

  // async DMA the 8 KB AO tile into LDS
  {
    const char* src = (const char*)(ao + ((size_t)(b * Nn) + i0) * Dn);
    unsigned base = lds_addr_of(&AOs[0][0]);
    async_g2l_b128(base + tid * 16,        src + tid * 16);
    async_g2l_b128(base + 4096 + tid * 16, src + 4096 + tid * 16);
  }
  wait_async0();
  __syncthreads();

  int nt0 = wave * 2, nt1 = wave * 2 + 1;
  const _Float16* b0p = WoutS + (size_t)nt0 * 512 + lane * 16;
  const _Float16* b1p = WoutS + (size_t)nt1 * 512 + lane * 16;
  v8f acc0 = {}, acc1 = {};
  v16h b0c = load_b_frag(b0p);
  v16h b1c = load_b_frag(b1p);
#pragma unroll
  for (int kc = 0; kc < 8; ++kc) {         // double-buffered weight stream
    v16h b0n = b0c, b1n = b1c;
    if (kc < 7) {
      b0n = load_b_frag(b0p + (size_t)(kc + 1) * 16 * 512);
      b1n = load_b_frag(b1p + (size_t)(kc + 1) * 16 * 512);
    }
    v16h a = load_a_frag(&AOs[l15][kc * 32], g);
    acc0 = __builtin_amdgcn_wmma_f32_16x16x32_f16(false, a, false, b0c, (short)0, acc0, false, false);
    acc1 = __builtin_amdgcn_wmma_f32_16x16x32_f16(false, a, false, b1c, (short)0, acc1, false, false);
    b0c = b0n; b1c = b1n;
  }
#pragma unroll
  for (int r = 0; r < 8; ++r) {
    int m = r + 8 * g;
    Y[m][nt0 * 16 + l15] = acc0[r];
    Y[m][nt1 * 16 + l15] = acc1[r];
  }
  __syncthreads();

  // residual + LayerNorm, 2 rows per wave
  for (int rr = 0; rr < 2; ++rr) {
    int row = wave * 2 + rr;
    int gi  = b * Nn + i0 + row;
    float sum = 0.f;
    for (int c = lane; c < 256; c += 32) {
      float y = Y[row][c] + x[(size_t)gi * Dn + c];
      Y[row][c] = y;
      sum += y;
    }
    sum = wave_sum(sum);
    float mu = sum * (1.0f / 256.0f);
    float vs = 0.f;
    for (int c = lane; c < 256; c += 32) { float d = Y[row][c] - mu; vs += d * d; }
    vs = wave_sum(vs);
    float rs = rsqrtf(vs * (1.0f / 256.0f) + EPSV);
    for (int c = lane; c < 256; c += 32)
      out[(size_t)gi * Dn + c] = (Y[row][c] - mu) * rs * gamma[c] + beta[c];
  }
}

// ---------------- host launch ----------------------------------------------
extern "C" void kernel_launch(void* const* d_in, const int* in_sizes, int n_in,
                              void* d_out, int out_size, void* d_ws, size_t ws_size,
                              hipStream_t stream) {
  (void)in_sizes; (void)n_in; (void)out_size; (void)ws_size;
  const float* x     = (const float*)d_in[0];
  const float* pos   = (const float*)d_in[1];
  const float* vel   = (const float*)d_in[2];
  const float* att   = (const float*)d_in[3];
  const float* Wqkv  = (const float*)d_in[4];
  const float* Wout  = (const float*)d_in[5];
  const float* gamma = (const float*)d_in[6];
  const float* beta  = (const float*)d_in[7];
  const float* relW  = (const float*)d_in[8];
  const float* relb  = (const float*)d_in[9];
  const int*   alive = (const int*)d_in[10];
  float* out = (float*)d_out;

  char* ws = (char*)d_ws;
  _Float16* WqkvS = (_Float16*)(ws);                               // 384 KB
  _Float16* WoutS = (_Float16*)(ws + 393216);                      // 128 KB
  float*    hdg   = (float*)(ws + 524288);                         // 128 KB
  _Float16* qh    = (_Float16*)(ws + 655360);                      // 8 MB
  _Float16* kh    = (_Float16*)(ws + 655360 + 8388608ull);         // 8 MB
  _Float16* vt    = (_Float16*)(ws + 655360 + 2ull * 8388608ull);  // 8 MB
  _Float16* ao    = (_Float16*)(ws + 655360 + 3ull * 8388608ull);  // 8 MB
  _Float16* xh    = (_Float16*)(ws + 655360 + 4ull * 8388608ull);  // 8 MB

  size_t attn_smem = 65536 + 32768 + 2048 + 192 * 4 + 64;  // ~101.2 KB of 320 KB LDS
  hipFuncSetAttribute((const void*)k_attn,
                      hipFuncAttributeMaxDynamicSharedMemorySize, (int)attn_smem);

  k_xh  <<<4096, 256, 0, stream>>>(x, xh);
  k_prep<<<1088, 256, 0, stream>>>(Wqkv, Wout, att, WqkvS, WoutS, hdg);
  k_qkv <<<Bn * (Nn / 16), 256, 0, stream>>>(xh, WqkvS, qh, kh, vt);
  k_attn<<<Bn * Hn * (Nn / 32), 256, attn_smem, stream>>>(qh, kh, vt, pos, vel, hdg,
                                                          relW, relb, alive, ao);
  k_proj<<<Bn * (Nn / 16), 256, 0, stream>>>(ao, WoutS, x, gamma, beta, out);
}
